// LocalContextEncoder_56195352100981
// MI455X (gfx1250) — matrix operand order repaired and checked by reading, not compiled, not run on hardware
//
#include <hip/hip_runtime.h>
#include <hip/hip_bf16.h>
#include <math.h>

#define DMODEL 768
#define LSEQ   256
#define NBATCH 4
#define HPAD   772   // 768 + 4 dwords: row stride padding to kill LDS bank conflicts

typedef __attribute__((ext_vector_type(16))) _Float16 v16h;
typedef __attribute__((ext_vector_type(8)))  float    v8f;
typedef __attribute__((ext_vector_type(4)))  int      v4i;

#define AS1 __attribute__((address_space(1)))
#define AS3 __attribute__((address_space(3)))

#if defined(__gfx1250__) && __has_builtin(__builtin_amdgcn_global_load_async_to_lds_b128) && __has_builtin(__builtin_amdgcn_s_wait_asynccnt)
#define GATE_ASYNC 1
#else
#define GATE_ASYNC 0
#endif

__device__ __forceinline__ v8f wmma_f16(v16h a, v16h b, v8f c) {
  // D = A(16x32 f16) * B(32x16 f16) + C(16x16 f32)
  return __builtin_amdgcn_wmma_f32_16x16x32_f16(false, a, false, b, (short)0, c,
                                                false, false);
}

// K-offset of the first f16 of dword v of the A fragment (contiguous k = 16*half + 2*v, matching B)
__device__ __forceinline__ int a_koff(int v, int half) {
  return half * 16 + 2 * v;
}
// K-offset of the first f16 of dword v of the B fragment (contiguous k = 16*half + 2*v, matching A)
__device__ __forceinline__ int b_koff(int v, int half) {
  return half * 16 + 2 * v;
}

// ---------------------------------------------------------------------------
// Kernel 1: hq = Q @ W1[:D] (+b1 folded in), hk = K @ W1[D:]
// One wave per 16x16 output tile. Rows flattened over B*L = 1024.
// ---------------------------------------------------------------------------
__global__ void __launch_bounds__(32)
gemm_hqhk(const float* __restrict__ Q, const float* __restrict__ Km,
          const float* __restrict__ W1, const float* __restrict__ b1,
          float* __restrict__ hq, float* __restrict__ hk) {
  const int ct = blockIdx.x;          // 0..47 (D cols / 16)
  const int rt = blockIdx.y;          // 0..63 (B*L rows / 16)
  const int z  = blockIdx.z;          // 0 -> hq, 1 -> hk
  const int l = threadIdx.x, half = l >> 4, mn = l & 15;

  const float* A  = z ? Km : Q;                      // [1024, 768]
  const float* Bm = W1 + (z ? DMODEL * DMODEL : 0);  // [768, 768]
  float*       C  = z ? hk : hq;

  const int arow = (rt * 16 + mn) * DMODEL;
  const int col  = ct * 16 + mn;
  const float bias = (z == 0) ? b1[col] : 0.0f;      // hoisted: one load, no epilogue branches

  v8f c = {};
  for (int kb = 0; kb < DMODEL; kb += 32) {
    v16h a, b;
#pragma unroll
    for (int v = 0; v < 8; ++v) {
      const int ka = kb + a_koff(v, half);
      float2 f = *(const float2*)(A + arow + ka);
      a[2 * v]     = (_Float16)f.x;
      a[2 * v + 1] = (_Float16)f.y;
      const int kk = kb + b_koff(v, half);
      b[2 * v]     = (_Float16)Bm[kk * DMODEL + col];
      b[2 * v + 1] = (_Float16)Bm[(kk + 1) * DMODEL + col];
    }
    c = wmma_f16(a, b, c);
  }
#pragma unroll
  for (int r = 0; r < 8; ++r) {
    const int m = r + 8 * half;
    C[(rt * 16 + m) * DMODEL + col] = c[r] + bias;
  }
}

// ---------------------------------------------------------------------------
// Kernel 2: S[b,i,j] = (Q[b,i,:] . K[b,j,:]) / sqrt(D)
// B fragment rows of K are contiguous in memory (K^T access is row reads).
// ---------------------------------------------------------------------------
__global__ void __launch_bounds__(32)
gemm_qk(const float* __restrict__ Q, const float* __restrict__ Km,
        float* __restrict__ S) {
  const int jt = blockIdx.x;          // 0..15
  const int it = blockIdx.y;          // 0..15
  const int b  = blockIdx.z;          // 0..3
  const int l = threadIdx.x, half = l >> 4, mn = l & 15;

  const float* Qb = Q  + b * LSEQ * DMODEL;
  const float* Kb = Km + b * LSEQ * DMODEL;

  v8f c = {};
  for (int kb = 0; kb < DMODEL; kb += 32) {
    v16h a, bb;
#pragma unroll
    for (int v = 0; v < 8; ++v) {
      const int ka = kb + a_koff(v, half);
      float2 f = *(const float2*)(Qb + (it * 16 + mn) * DMODEL + ka);
      a[2 * v]     = (_Float16)f.x;
      a[2 * v + 1] = (_Float16)f.y;
      const int kk = kb + b_koff(v, half);
      float2 g = *(const float2*)(Kb + (jt * 16 + mn) * DMODEL + kk);
      bb[2 * v]     = (_Float16)g.x;
      bb[2 * v + 1] = (_Float16)g.y;
    }
    c = wmma_f16(a, bb, c);
  }
  const float scale = 0.03608439182435161f;  // 1/sqrt(768)
#pragma unroll
  for (int r = 0; r < 8; ++r) {
    const int m = r + 8 * half;
    S[(b * LSEQ + it * 16 + m) * LSEQ + jt * 16 + mn] = c[r] * scale;
  }
}

// ---------------------------------------------------------------------------
// Kernel 3: gated = S * sigmoid( sum_d relu(hq[i,d]+hk[j,d]) * W2[d] + b2 )
// One WG per 16x16 (i,j) tile; hq/hk tiles staged in LDS (async direct
// global->LDS path on gfx1250, ASYNCcnt-tracked, bypassing VGPRs).
// This is the dominant (non-matrix-izable) VALU work: B*L*L*D relu-fmas.
// In-place update of S -> gated scores.
// ---------------------------------------------------------------------------
__global__ void __launch_bounds__(256)
gate_kernel(const float* __restrict__ hq, const float* __restrict__ hk,
            const float* __restrict__ W2, const float* __restrict__ b2,
            const int* __restrict__ mask, float* __restrict__ S) {
  __shared__ float sq[16 * HPAD];
  __shared__ float sk[16 * HPAD];
  __shared__ float sw[DMODEL];

  const int jt = blockIdx.x, it = blockIdx.y, b = blockIdx.z;
  const int t = threadIdx.x;

  const float4* gq = (const float4*)(hq + (b * LSEQ + it * 16) * DMODEL);
  const float4* gk = (const float4*)(hk + (b * LSEQ + jt * 16) * DMODEL);
  for (int p = t; p < 16 * (DMODEL / 4); p += 256) {
    const int row = p / (DMODEL / 4), c4 = p % (DMODEL / 4);
#if GATE_ASYNC
    __builtin_amdgcn_global_load_async_to_lds_b128(
        (v4i*)(gq + row * (DMODEL / 4) + c4),
        (AS3 v4i*)(&sq[row * HPAD + c4 * 4]), 0, 0);
    __builtin_amdgcn_global_load_async_to_lds_b128(
        (v4i*)(gk + row * (DMODEL / 4) + c4),
        (AS3 v4i*)(&sk[row * HPAD + c4 * 4]), 0, 0);
#else
    *(float4*)&sq[row * HPAD + c4 * 4] = gq[row * (DMODEL / 4) + c4];
    *(float4*)&sk[row * HPAD + c4 * 4] = gk[row * (DMODEL / 4) + c4];
#endif
  }
  for (int p = t; p < DMODEL; p += 256) sw[p] = W2[p];
#if GATE_ASYNC
  __builtin_amdgcn_s_wait_asynccnt(0);
#endif
  __syncthreads();

  const int i = t >> 4, j = t & 15;
  const float* rq = &sq[i * HPAD];
  const float* rk = &sk[j * HPAD];
  float acc = 0.f;
  for (int d = 0; d < DMODEL; d += 4) {
    float4 a = *(const float4*)(rq + d);
    float4 k = *(const float4*)(rk + d);
    float4 w = *(const float4*)(&sw[d]);
    acc += fmaxf(a.x + k.x, 0.f) * w.x;
    acc += fmaxf(a.y + k.y, 0.f) * w.y;
    acc += fmaxf(a.z + k.z, 0.f) * w.z;
    acc += fmaxf(a.w + k.w, 0.f) * w.w;
  }
  const float conflict = acc + b2[0];
  const int gi = (b * LSEQ + it * 16 + i) * LSEQ + jt * 16 + j;
  float g = S[gi] * (1.f / (1.f + __expf(-conflict)));
  if (mask[gi] == 0) g = -INFINITY;
  S[gi] = g;
}

// ---------------------------------------------------------------------------
// Kernel 4: row softmax over j; writes attn_weights into d_out tail.
// One 256-thread WG per (b,i) row.
// ---------------------------------------------------------------------------
__global__ void __launch_bounds__(256)
softmax_kernel(const float* __restrict__ G, float* __restrict__ Wout) {
  __shared__ float red[256];
  const int row = blockIdx.x;   // 0..1023
  const int t = threadIdx.x;
  const float g = G[row * LSEQ + t];
  red[t] = g;
  __syncthreads();
  for (int s = 128; s > 0; s >>= 1) {
    if (t < s) red[t] = fmaxf(red[t], red[t + s]);
    __syncthreads();
  }
  const float mx = red[0];
  __syncthreads();
  const float e = __expf(g - mx);
  red[t] = e;
  __syncthreads();
  for (int s = 128; s > 0; s >>= 1) {
    if (t < s) red[t] += red[t + s];
    __syncthreads();
  }
  Wout[row * LSEQ + t] = e / red[0];
}

// ---------------------------------------------------------------------------
// Kernel 5: out = P @ V   ([256,256] x [256,768] per batch)
// ---------------------------------------------------------------------------
__global__ void __launch_bounds__(32)
gemm_pv(const float* __restrict__ P, const float* __restrict__ V,
        float* __restrict__ Out) {
  const int ct = blockIdx.x;          // 0..47
  const int rt = blockIdx.y;          // 0..15
  const int b  = blockIdx.z;          // 0..3
  const int l = threadIdx.x, half = l >> 4, mn = l & 15;

  const float* Pb = P + b * LSEQ * LSEQ;
  const float* Vb = V + b * LSEQ * DMODEL;
  const int col = ct * 16 + mn;

  v8f c = {};
  for (int kb = 0; kb < LSEQ; kb += 32) {
    v16h a, bb;
#pragma unroll
    for (int v = 0; v < 8; ++v) {
      const int ka = kb + a_koff(v, half);
      float2 f = *(const float2*)(Pb + (rt * 16 + mn) * LSEQ + ka);
      a[2 * v]     = (_Float16)f.x;
      a[2 * v + 1] = (_Float16)f.y;
      const int kk = kb + b_koff(v, half);
      bb[2 * v]     = (_Float16)Vb[kk * DMODEL + col];
      bb[2 * v + 1] = (_Float16)Vb[(kk + 1) * DMODEL + col];
    }
    c = wmma_f16(a, bb, c);
  }
#pragma unroll
  for (int r = 0; r < 8; ++r) {
    const int m = r + 8 * half;
    Out[(b * LSEQ + rt * 16 + m) * DMODEL + col] = c[r];
  }
}

// ---------------------------------------------------------------------------
extern "C" void kernel_launch(void* const* d_in, const int* in_sizes, int n_in,
                              void* d_out, int out_size, void* d_ws, size_t ws_size,
                              hipStream_t stream) {
  const float* Q    = (const float*)d_in[0];
  const float* Kmat = (const float*)d_in[1];
  const float* V    = (const float*)d_in[2];
  const int*   mask = (const int*)d_in[3];
  const float* W1   = (const float*)d_in[4];
  const float* b1   = (const float*)d_in[5];
  const float* W2   = (const float*)d_in[6];
  const float* b2   = (const float*)d_in[7];

  float* out  = (float*)d_out;                    // [B,L,D]
  float* attn = out + NBATCH * LSEQ * DMODEL;     // [B,L,L]

  float* hq = (float*)d_ws;                       // [B*L, D]  3 MB
  float* hk = hq + NBATCH * LSEQ * DMODEL;        // [B*L, D]  3 MB
  float* S  = hk + NBATCH * LSEQ * DMODEL;        // [B,L,L]   1 MB

  gemm_hqhk<<<dim3(48, 64, 2), 32, 0, stream>>>(Q, Kmat, W1, b1, hq, hk);
  gemm_qk<<<dim3(16, 16, 4), 32, 0, stream>>>(Q, Kmat, S);
  gate_kernel<<<dim3(16, 16, 4), 256, 0, stream>>>(hq, hk, W2, b2, mask, S);
  softmax_kernel<<<NBATCH * LSEQ, 256, 0, stream>>>(S, attn);
  gemm_pv<<<dim3(48, 16, 4), 32, 0, stream>>>(attn, V, out);
}
